// MoEBlock_29265907155520
// MI455X (gfx1250) — matrix operand order repaired and checked by reading, not compiled, not run on hardware
//
#include <hip/hip_runtime.h>
#include <cmath>

#define T_TOK   8192
#define DIM_D   1024
#define HID     2048
#define NEXP    8
#define TOPK    2
#define CAP     8192

typedef __attribute__((ext_vector_type(16))) __bf16 bf16x16;
typedef __attribute__((ext_vector_type(8)))  float  f32x8;

union FragU { uint4 q[2]; bf16x16 v; };

__device__ __forceinline__ unsigned short f2bf(float f) {
  unsigned int u = __float_as_uint(f);
  unsigned int r = u + 0x7FFFu + ((u >> 16) & 1u);
  return (unsigned short)(r >> 16);
}

__device__ __forceinline__ f32x8 wmma_bf16(bf16x16 a, bf16x16 b, f32x8 c) {
  return __builtin_amdgcn_wmma_f32_16x16x32_bf16(false, a, false, b, (short)0, c,
                                                 false, false);
}

// Async DMA: copy 2x16B from global to LDS (both sides get INST_OFFSET),
// tracked by ASYNCcnt; data lands in LDS without touching VGPRs.
__device__ __forceinline__ void async_copy32(unsigned int lds_addr,
                                             unsigned long long gaddr) {
  asm volatile(
      "global_load_async_to_lds_b128 %0, %1, off\n\t"
      "global_load_async_to_lds_b128 %0, %1, off offset:16"
      :: "v"(lds_addr), "v"(gaddr) : "memory");
}

__device__ __forceinline__ void wait_async_all() {
  asm volatile("s_wait_asynccnt 0x0" ::: "memory");
}

// A fragment: 16(M) x 32(K) bf16 tile from LDS, tile rows stride = 32 halves.
// Lane L: m = L&15, g = L>>4; halves [0..7] = A[m][g*8+0..7],
//         halves [8..15] = A[m][16+g*8+0..7]  (ISA 16-bit A layout).
__device__ __forceinline__ bf16x16 load_frag_A(const unsigned short* base, int row0, int lane) {
  int g = lane >> 4, m = lane & 15;
  const unsigned short* p = base + (row0 + m) * 32;
  FragU u;
  u.q[0] = *(const uint4*)(p + (g << 3));
  u.q[1] = *(const uint4*)(p + 16 + (g << 3));
  return u.v;
}

// B fragment: 32(K) x 16(N); staged n-major [n][32 halves]. Lane L: n = L&15,
// halves [0..7] = K (L>>4)*8+0..7, halves [8..15] = K 16+(L>>4)*8+0..7.
__device__ __forceinline__ bf16x16 load_frag_B(const unsigned short* base, int n0, int lane) {
  int g = lane >> 4, n = lane & 15;
  const unsigned short* p = base + (n0 + n) * 32;
  FragU u;
  u.q[0] = *(const uint4*)(p + (g << 3));
  u.q[1] = *(const uint4*)(p + 16 + (g << 3));
  return u.v;
}

__global__ void zero_kernel(float* out, int n, int* cnts) {
  int i = blockIdx.x * blockDim.x + threadIdx.x;
  if (i < n) out[i] = 0.0f;
  if (i < NEXP) cnts[i] = 0;
}

__global__ void cvt_bf16_kernel(const float* __restrict__ s, unsigned short* __restrict__ d, int n) {
  int stride = gridDim.x * blockDim.x * 4;
  for (int i = (blockIdx.x * blockDim.x + threadIdx.x) * 4; i < n; i += stride) {
    float4 f = *(const float4*)(s + i);
    uint2 o;
    o.x = (unsigned int)f2bf(f.x) | ((unsigned int)f2bf(f.y) << 16);
    o.y = (unsigned int)f2bf(f.z) | ((unsigned int)f2bf(f.w) << 16);
    *(uint2*)(d + i) = o;
  }
}

__global__ void __launch_bounds__(256) router_kernel(
    const float* __restrict__ x, const float* __restrict__ rw,
    float* __restrict__ probs, int* __restrict__ cnts,
    int* __restrict__ list, float* __restrict__ gatev) {
  int wave = threadIdx.x >> 5;
  int lane = threadIdx.x & 31;
  int t = blockIdx.x * 8 + wave;
  float acc[NEXP];
#pragma unroll
  for (int e = 0; e < NEXP; ++e) acc[e] = 0.f;
  const float* xr = x + (size_t)t * DIM_D;
  for (int i = lane; i < DIM_D; i += 32) {
    float xv = xr[i];
#pragma unroll
    for (int e = 0; e < NEXP; ++e) acc[e] += xv * rw[e * DIM_D + i];
  }
#pragma unroll
  for (int e = 0; e < NEXP; ++e) {
#pragma unroll
    for (int off = 16; off > 0; off >>= 1)
      acc[e] += __shfl_xor(acc[e], off, 32);
  }
  if (lane == 0) {
    float mx = acc[0];
#pragma unroll
    for (int e = 1; e < NEXP; ++e) mx = fmaxf(mx, acc[e]);
    float pe[NEXP];
    float s = 0.f;
#pragma unroll
    for (int e = 0; e < NEXP; ++e) { pe[e] = expf(acc[e] - mx); s += pe[e]; }
    float inv = 1.f / s;
#pragma unroll
    for (int e = 0; e < NEXP; ++e) probs[t * NEXP + e] = pe[e] * inv;
    // top-2 (strict > keeps earliest index, matching lax.top_k tie-break)
    int e0 = 0; float l0 = acc[0];
#pragma unroll
    for (int e = 1; e < NEXP; ++e) if (acc[e] > l0) { l0 = acc[e]; e0 = e; }
    int e1 = (e0 == 0) ? 1 : 0; float l1 = acc[e1];
#pragma unroll
    for (int e = 0; e < NEXP; ++e)
      if (e != e0 && acc[e] > l1) { l1 = acc[e]; e1 = e; }
    float q = expf(l1 - l0);
    float g0 = 1.f / (1.f + q);
    float g1 = q / (1.f + q);
    int p0 = atomicAdd(&cnts[e0], 1);
    list[e0 * CAP + p0] = t; gatev[e0 * CAP + p0] = g0;
    int p1 = atomicAdd(&cnts[e1], 1);
    list[e1 * CAP + p1] = t; gatev[e1 * CAP + p1] = g1;
  }
}

__global__ void offsets_kernel(const int* __restrict__ cnts, int* __restrict__ offs) {
  if (threadIdx.x == 0 && blockIdx.x == 0) {
    int s = 0;
    for (int e = 0; e < NEXP; ++e) { offs[e] = s; s += cnts[e]; }
    offs[NEXP] = s;
  }
}

// H[slot][0..2047] = gelu(X[tok[slot]] @ w1[e]^T + b1[e]) for gathered slots.
__global__ void __launch_bounds__(256) gemm1_kernel(
    const unsigned short* __restrict__ xb, const unsigned short* __restrict__ w1b,
    const float* __restrict__ b1, const int* __restrict__ list,
    const int* __restrict__ cnts, const int* __restrict__ offs,
    unsigned short* __restrict__ H) {
  const int e = blockIdx.z;
  const int nb = blockIdx.y;
  const int tile = blockIdx.x;
  const int cnt = cnts[e];
  if (tile * 128 >= cnt) return;
  const int hbase = offs[e];

  __shared__ __align__(16) unsigned short Xs[2][128 * 32];
  __shared__ __align__(16) unsigned short Ws[2][128 * 32];
  __shared__ int rowTok[128];

  const int tid = threadIdx.x;
  const int lane = tid & 31;
  const int wv = tid >> 5;
  const int mw = wv & 1;   // M half: rows mw*64 .. +63
  const int nw = wv >> 1;  // N quarter: cols nw*32 .. +31

  if (tid < 128) {
    int slot = tile * 128 + tid;
    int sc = slot < cnt ? slot : cnt - 1;
    rowTok[tid] = list[e * CAP + sc];
  }
  __syncthreads();

  const int sr = tid >> 1;   // staged row 0..127
  const int sh = tid & 1;    // 32B half of the 64B row chunk
  const unsigned short* xsrc_base = xb + (size_t)rowTok[sr] * DIM_D + sh * 16;
  const unsigned short* wsrc_base =
      w1b + ((size_t)e * HID + (size_t)nb * 128 + sr) * DIM_D + sh * 16;
  const unsigned int xlds0 = (unsigned int)(size_t)&Xs[0][sr * 32 + sh * 16];
  const unsigned int xlds1 = (unsigned int)(size_t)&Xs[1][sr * 32 + sh * 16];
  const unsigned int wlds0 = (unsigned int)(size_t)&Ws[0][sr * 32 + sh * 16];
  const unsigned int wlds1 = (unsigned int)(size_t)&Ws[1][sr * 32 + sh * 16];

  auto STAGE = [&](int kt, int buf) {
    async_copy32(buf ? xlds1 : xlds0,
                 (unsigned long long)(size_t)(xsrc_base + kt * 32));
    async_copy32(buf ? wlds1 : wlds0,
                 (unsigned long long)(size_t)(wsrc_base + kt * 32));
  };

  f32x8 accv[4][2];
#pragma unroll
  for (int mi = 0; mi < 4; ++mi)
#pragma unroll
    for (int ni = 0; ni < 2; ++ni)
#pragma unroll
      for (int j = 0; j < 8; ++j) accv[mi][ni][j] = 0.f;

  STAGE(0, 0);
  wait_async_all();
  __syncthreads();

  const int KT = DIM_D / 32;
  for (int kt = 0; kt < KT; ++kt) {
    const int cur = kt & 1;
    if (kt + 1 < KT) STAGE(kt + 1, cur ^ 1);  // DMA next tile while computing
    bf16x16 afr[4], bfr[2];
#pragma unroll
    for (int mi = 0; mi < 4; ++mi)
      afr[mi] = load_frag_A(&Xs[cur][0], mw * 64 + mi * 16, lane);
#pragma unroll
    for (int ni = 0; ni < 2; ++ni)
      bfr[ni] = load_frag_B(&Ws[cur][0], nw * 32 + ni * 16, lane);
#pragma unroll
    for (int mi = 0; mi < 4; ++mi)
#pragma unroll
      for (int ni = 0; ni < 2; ++ni)
        accv[mi][ni] = wmma_bf16(afr[mi], bfr[ni], accv[mi][ni]);
    wait_async_all();
    __syncthreads();
  }

  const int g = lane >> 4;
  const int cl = lane & 15;
#pragma unroll
  for (int mi = 0; mi < 4; ++mi) {
#pragma unroll
    for (int ni = 0; ni < 2; ++ni) {
      int ngl = nb * 128 + nw * 32 + ni * 16 + cl;
      float bias = b1[e * HID + ngl];
#pragma unroll
      for (int r = 0; r < 8; ++r) {
        int rowl = mw * 64 + mi * 16 + g * 8 + r;
        int slot = tile * 128 + rowl;
        if (slot < cnt) {
          float v = accv[mi][ni][r] + bias;
          float gl = 0.5f * v * (1.0f + erff(v * 0.70710678118654752440f));
          H[(size_t)(hbase + slot) * HID + ngl] = f2bf(gl);
        }
      }
    }
  }
}

// out[tok] += gate * (H[slot] @ w2[e]^T + b2[e])
__global__ void __launch_bounds__(256) gemm2_kernel(
    const unsigned short* __restrict__ H, const unsigned short* __restrict__ w2b,
    const float* __restrict__ b2, const int* __restrict__ list,
    const float* __restrict__ gatev, const int* __restrict__ cnts,
    const int* __restrict__ offs, float* __restrict__ out) {
  const int e = blockIdx.z;
  const int nb = blockIdx.y;
  const int tile = blockIdx.x;
  const int cnt = cnts[e];
  if (tile * 128 >= cnt) return;
  const int hbase = offs[e];

  __shared__ __align__(16) unsigned short Xs[2][128 * 32];
  __shared__ __align__(16) unsigned short Ws[2][128 * 32];
  __shared__ int rowTok[128];
  __shared__ float rowGate[128];

  const int tid = threadIdx.x;
  const int lane = tid & 31;
  const int wv = tid >> 5;
  const int mw = wv & 1;
  const int nw = wv >> 1;

  if (tid < 128) {
    int slot = tile * 128 + tid;
    if (slot < cnt) {
      rowTok[tid] = list[e * CAP + slot];
      rowGate[tid] = gatev[e * CAP + slot];
    } else {
      rowTok[tid] = 0;
      rowGate[tid] = 0.f;
    }
  }
  __syncthreads();

  const int sr = tid >> 1;
  const int sh = tid & 1;
  int slotc = tile * 128 + sr;
  if (slotc >= cnt) slotc = cnt - 1;
  const unsigned short* xsrc_base = H + (size_t)(hbase + slotc) * HID + sh * 16;
  const unsigned short* wsrc_base =
      w2b + ((size_t)e * DIM_D + (size_t)nb * 128 + sr) * HID + sh * 16;
  const unsigned int xlds0 = (unsigned int)(size_t)&Xs[0][sr * 32 + sh * 16];
  const unsigned int xlds1 = (unsigned int)(size_t)&Xs[1][sr * 32 + sh * 16];
  const unsigned int wlds0 = (unsigned int)(size_t)&Ws[0][sr * 32 + sh * 16];
  const unsigned int wlds1 = (unsigned int)(size_t)&Ws[1][sr * 32 + sh * 16];

  auto STAGE = [&](int kt, int buf) {
    async_copy32(buf ? xlds1 : xlds0,
                 (unsigned long long)(size_t)(xsrc_base + kt * 32));
    async_copy32(buf ? wlds1 : wlds0,
                 (unsigned long long)(size_t)(wsrc_base + kt * 32));
  };

  f32x8 accv[4][2];
#pragma unroll
  for (int mi = 0; mi < 4; ++mi)
#pragma unroll
    for (int ni = 0; ni < 2; ++ni)
#pragma unroll
      for (int j = 0; j < 8; ++j) accv[mi][ni][j] = 0.f;

  STAGE(0, 0);
  wait_async_all();
  __syncthreads();

  const int KT = HID / 32;
  for (int kt = 0; kt < KT; ++kt) {
    const int cur = kt & 1;
    if (kt + 1 < KT) STAGE(kt + 1, cur ^ 1);
    bf16x16 afr[4], bfr[2];
#pragma unroll
    for (int mi = 0; mi < 4; ++mi)
      afr[mi] = load_frag_A(&Xs[cur][0], mw * 64 + mi * 16, lane);
#pragma unroll
    for (int ni = 0; ni < 2; ++ni)
      bfr[ni] = load_frag_B(&Ws[cur][0], nw * 32 + ni * 16, lane);
#pragma unroll
    for (int mi = 0; mi < 4; ++mi)
#pragma unroll
      for (int ni = 0; ni < 2; ++ni)
        accv[mi][ni] = wmma_bf16(afr[mi], bfr[ni], accv[mi][ni]);
    wait_async_all();
    __syncthreads();
  }

  const int g = lane >> 4;
  const int cl = lane & 15;
#pragma unroll
  for (int mi = 0; mi < 4; ++mi) {
#pragma unroll
    for (int ni = 0; ni < 2; ++ni) {
      int ngl = nb * 128 + nw * 32 + ni * 16 + cl;
      float bias = b2[e * DIM_D + ngl];
#pragma unroll
      for (int r = 0; r < 8; ++r) {
        int rowl = mw * 64 + mi * 16 + g * 8 + r;
        int slot = tile * 128 + rowl;
        if (slot < cnt) {
          float y = accv[mi][ni][r] + bias;
          atomicAdd(&out[(size_t)rowTok[rowl] * DIM_D + ngl], rowGate[rowl] * y);
        }
      }
    }
  }
}

__global__ void __launch_bounds__(256) aux_kernel(const float* __restrict__ probs,
                                                  const int* __restrict__ cnts,
                                                  float* __restrict__ out_aux) {
  __shared__ float red[NEXP][256];
  __shared__ float term[NEXP];
  float p[NEXP];
#pragma unroll
  for (int e = 0; e < NEXP; ++e) p[e] = 0.f;
  for (int t = threadIdx.x; t < T_TOK; t += 256) {
#pragma unroll
    for (int e = 0; e < NEXP; ++e) p[e] += probs[t * NEXP + e];
  }
#pragma unroll
  for (int e = 0; e < NEXP; ++e) red[e][threadIdx.x] = p[e];
  __syncthreads();
  if (threadIdx.x < NEXP) {
    float s = 0.f;
    for (int i = 0; i < 256; ++i) s += red[threadIdx.x][i];
    float P = s / (float)T_TOK;
    float f = (float)cnts[threadIdx.x] / (float)(T_TOK * TOPK);
    term[threadIdx.x] = f * P;
  }
  __syncthreads();
  if (threadIdx.x == 0) {
    float a = 0.f;
    for (int e = 0; e < NEXP; ++e) a += term[e];
    *out_aux = (float)NEXP * a;
  }
}

extern "C" void kernel_launch(void* const* d_in, const int* in_sizes, int n_in,
                              void* d_out, int out_size, void* d_ws, size_t ws_size,
                              hipStream_t stream) {
  const float* x  = (const float*)d_in[0];
  const float* rw = (const float*)d_in[1];
  const float* w1 = (const float*)d_in[2];
  const float* b1 = (const float*)d_in[3];
  const float* w2 = (const float*)d_in[4];
  const float* b2 = (const float*)d_in[5];
  float* out = (float*)d_out;

  // workspace carve (bf16 buffers then metadata); ~152 MB total
  unsigned short* xb  = (unsigned short*)d_ws;
  unsigned short* w1b = xb + (size_t)T_TOK * DIM_D;
  unsigned short* w2b = w1b + (size_t)NEXP * HID * DIM_D;
  unsigned short* Hb  = w2b + (size_t)NEXP * DIM_D * HID;
  float* probs = (float*)(Hb + (size_t)T_TOK * TOPK * HID);
  int* cnts = (int*)(probs + (size_t)T_TOK * NEXP);
  int* offs = cnts + 16;
  int* list = offs + 16;
  float* gatev = (float*)(list + NEXP * CAP);

  zero_kernel<<<(out_size + 255) / 256, 256, 0, stream>>>(out, out_size, cnts);
  cvt_bf16_kernel<<<2048, 256, 0, stream>>>(x,  xb,  T_TOK * DIM_D);
  cvt_bf16_kernel<<<2048, 256, 0, stream>>>(w1, w1b, NEXP * HID * DIM_D);
  cvt_bf16_kernel<<<2048, 256, 0, stream>>>(w2, w2b, NEXP * DIM_D * HID);
  router_kernel<<<T_TOK / 8, 256, 0, stream>>>(x, rw, probs, cnts, list, gatev);
  offsets_kernel<<<1, 32, 0, stream>>>(cnts, offs);
  gemm1_kernel<<<dim3(CAP / 128, HID / 128, NEXP), 256, 0, stream>>>(
      xb, w1b, b1, list, cnts, offs, Hb);
  gemm2_kernel<<<dim3(CAP / 128, DIM_D / 128, NEXP), 256, 0, stream>>>(
      Hb, w2b, b2, list, gatev, cnts, offs, out);
  aux_kernel<<<1, 256, 0, stream>>>(probs, cnts, out + (out_size - 1));
}
